// ReadNetBlock_33913061769388
// MI455X (gfx1250) — compile-verified, matrix operand order checked by reading, not run on hardware
//
#include <hip/hip_runtime.h>
#include <hip/hip_bf16.h>
#include <math.h>

typedef unsigned short u16;
typedef unsigned int   u32;

typedef __attribute__((ext_vector_type(16))) __bf16 v16bf;
typedef __attribute__((ext_vector_type(8)))  float  v8f;

// CDNA5 async global->LDS path (ASYNCcnt-tracked), with sync fallback.
// Probe round 2 revealed the builtin signature wants int4 pointers:
//   void __builtin_amdgcn_global_load_async_to_lds_b128(int4 AS1*, int4 AS3*, Ii, Ii)
#if defined(__AMDGCN__) && defined(__gfx1250__) && \
    __has_builtin(__builtin_amdgcn_global_load_async_to_lds_b128) && \
    __has_builtin(__builtin_amdgcn_s_wait_asynccnt)
#define USE_ASYNC_LDS 1
typedef __attribute__((ext_vector_type(4))) int i4;
typedef __attribute__((address_space(1))) i4 g_i4;
typedef __attribute__((address_space(3))) i4 l_i4;
#else
#define USE_ASYNC_LDS 0
#endif

union FragBF {
    v16bf v;
    uint4 q[2];
    u16   e[16];
};

__device__ __forceinline__ u16 f2bf(float f) {
    u32 u = __float_as_uint(f);
    u32 r = u + 0x7FFFu + ((u >> 16) & 1u);   // round-to-nearest-even
    return (u16)(r >> 16);
}

// ---------------------------------------------------------------------------
// Weight conversion kernels
// ---------------------------------------------------------------------------
__global__ __launch_bounds__(256)
void fuse_qkv_kernel(const float* __restrict__ wq, const float* __restrict__ wk,
                     const float* __restrict__ wv, u16* __restrict__ o) {
    // o: [L][2304][768] bf16 ; rows 0..767=wq, 768..1535=wk, 1536..2303=wv
    size_t i = (size_t)blockIdx.x * 256 + threadIdx.x;   // < 4*2304*768
    const size_t per = 2304ull * 768ull;
    int l = (int)(i / per);
    size_t r = i % per;
    int row = (int)(r / 768), k = (int)(r % 768);
    const float* src; int rr;
    if (row < 768)        { src = wq; rr = row; }
    else if (row < 1536)  { src = wk; rr = row - 768; }
    else                  { src = wv; rr = row - 1536; }
    o[i] = f2bf(src[((size_t)l * 768 + rr) * 768 + k]);
}

__global__ __launch_bounds__(256)
void f32_to_bf16_kernel(const float* __restrict__ a, u16* __restrict__ o, int n) {
    int i = blockIdx.x * 256 + threadIdx.x;
    if (i < n) o[i] = f2bf(a[i]);
}

// ---------------------------------------------------------------------------
// Positional encoding: y = x + enc, write fp32 + bf16
// ---------------------------------------------------------------------------
__global__ __launch_bounds__(256)
void posenc_kernel(const float* __restrict__ x, float* __restrict__ y,
                   u16* __restrict__ yh) {
    const int row = blockIdx.x;          // b*S + s
    const int s   = row & 511;
    const int tid = threadIdx.x;
    const size_t base = (size_t)row * 768;
    #pragma unroll
    for (int i = 0; i < 3; i++) {
        int d = tid + i * 256;
        float ex   = (float)(2 * (d >> 1)) * (1.0f / 768.0f);
        float freq = __powf(10000.0f, -ex);
        float ang  = (float)s * freq;
        float e    = (d & 1) ? __cosf(ang) : __sinf(ang);
        float o    = x[base + d] + e;
        y[base + d]  = o;
        yh[base + d] = f2bf(o);
    }
}

// ---------------------------------------------------------------------------
// Tiled bf16 WMMA GEMM:  C[M,N] = act(A[M,K] @ W[N,K]^T + bias)
// block tile 128x128, 8 waves (4 M x 2 N), wave tile 32x64, K-steps of 32.
// Double-buffered LDS, async global->LDS prefetch of tile t+1 during tile t.
// ACT: 0=none 1=relu 2=tanh
// ---------------------------------------------------------------------------
template <int ACT, bool OUT_BF16>
__global__ __launch_bounds__(256)
void gemm_bf16_wmma(const u16* __restrict__ A, const u16* __restrict__ W,
                    const float* __restrict__ bias,
                    float* __restrict__ outF, u16* __restrict__ outH,
                    int M, int N, int K) {
    __shared__ __align__(16) u16 As[2][128][32];
    __shared__ __align__(16) u16 Bs[2][128][32];

    const int tid  = threadIdx.x;
    const int lane = tid & 31;
    const int wid  = tid >> 5;
    const int wm   = wid & 3;            // 0..3  (M)
    const int wn   = wid >> 2;           // 0..1  (N)
    const int m0   = blockIdx.y * 128;
    const int n0   = blockIdx.x * 128;

    const int lrow = lane & 15;          // row/col within 16
    const int hi   = lane >> 4;          // half-wave
    const int kc0  = hi * 8;             // per-lane K chunk base

    v8f acc[2][4] = {};

    const int ldr = tid >> 1;            // 0..127 tile row
    const int ldc = (tid & 1) * 16;      // 0 or 16
    const u16* gA = A + (size_t)(m0 + ldr) * K + ldc;
    const u16* gW = W + (size_t)(n0 + ldr) * K + ldc;
    const int NKT = K / 32;

#if USE_ASYNC_LDS
    auto stage = [&](int kt, int buf) {
        __builtin_amdgcn_global_load_async_to_lds_b128(
            (g_i4*)(gA + kt),     (l_i4*)&As[buf][ldr][ldc],     0, 0);
        __builtin_amdgcn_global_load_async_to_lds_b128(
            (g_i4*)(gA + kt + 8), (l_i4*)&As[buf][ldr][ldc + 8], 0, 0);
        __builtin_amdgcn_global_load_async_to_lds_b128(
            (g_i4*)(gW + kt),     (l_i4*)&Bs[buf][ldr][ldc],     0, 0);
        __builtin_amdgcn_global_load_async_to_lds_b128(
            (g_i4*)(gW + kt + 8), (l_i4*)&Bs[buf][ldr][ldc + 8], 0, 0);
    };
#else
    auto stage = [&](int kt, int buf) {
        const uint4* ga = (const uint4*)(gA + kt);
        const uint4* gb = (const uint4*)(gW + kt);
        *(uint4*)&As[buf][ldr][ldc]     = ga[0];
        *(uint4*)&As[buf][ldr][ldc + 8] = ga[1];
        *(uint4*)&Bs[buf][ldr][ldc]     = gb[0];
        *(uint4*)&Bs[buf][ldr][ldc + 8] = gb[1];
    };
#endif

    auto compute = [&](int buf) {
        FragBF a[2], b[4];
        #pragma unroll
        for (int i = 0; i < 2; i++) {
            int r = wm * 32 + i * 16 + lrow;
            a[i].q[0] = *(const uint4*)&As[buf][r][kc0];
            a[i].q[1] = *(const uint4*)&As[buf][r][kc0 + 16];
        }
        #pragma unroll
        for (int j = 0; j < 4; j++) {
            int c = wn * 64 + j * 16 + lrow;
            b[j].q[0] = *(const uint4*)&Bs[buf][c][kc0];
            b[j].q[1] = *(const uint4*)&Bs[buf][c][kc0 + 16];
        }
        #pragma unroll
        for (int i = 0; i < 2; i++)
            #pragma unroll
            for (int j = 0; j < 4; j++)
                acc[i][j] = __builtin_amdgcn_wmma_f32_16x16x32_bf16(
                    false, a[i].v, false, b[j].v, (short)0, acc[i][j],
                    false, false);
    };

#if USE_ASYNC_LDS
    stage(0, 0);                                  // 4 async ops in flight
    for (int t = 0; t < NKT; t++) {
        const int buf = t & 1;
        if (t + 1 < NKT) {
            stage((t + 1) * 32, buf ^ 1);         // 8 in flight
            __builtin_amdgcn_s_wait_asynccnt(4);  // oldest 4 (tile t) done
        } else {
            __builtin_amdgcn_s_wait_asynccnt(0);
        }
        __syncthreads();
        compute(buf);
        __syncthreads();
    }
#else
    for (int t = 0; t < NKT; t++) {
        const int buf = t & 1;
        stage(t * 32, buf);
        __syncthreads();
        compute(buf);
        __syncthreads();
    }
#endif

    #pragma unroll
    for (int i = 0; i < 2; i++) {
        int mrow = m0 + wm * 32 + i * 16 + hi * 8;   // + r below
        #pragma unroll
        for (int j = 0; j < 4; j++) {
            int ncol = n0 + wn * 64 + j * 16 + lrow;
            float bz = bias ? bias[ncol] : 0.0f;
            #pragma unroll
            for (int r = 0; r < 8; r++) {
                float v = acc[i][j][r] + bz;
                if (ACT == 1) v = fmaxf(v, 0.0f);
                if (ACT == 2) v = tanhf(v);
                size_t o = (size_t)(mrow + r) * N + ncol;
                if (OUT_BF16) outH[o] = f2bf(v);
                else          outF[o] = v;
            }
        }
    }
}

// ---------------------------------------------------------------------------
// Fused attention: one block per (query-block of 32, head, batch).
// qkv: [B,S,3D] bf16 (q|k|v per row). out: [B,S,D] fp32.
// LDS: Qs 4KB | Sc 64KB | Pb 32KB | Vs[2] 8KB | red 1KB  (~110KB dynamic)
// ---------------------------------------------------------------------------
__global__ __launch_bounds__(256)
void attn_wmma(const u16* __restrict__ qkv, float* __restrict__ out) {
    constexpr int Sq = 32, HD = 64, Sk = 512, TD = 2304, Dm = 768;
    extern __shared__ char smem[];
    u16   (*Qs)[HD]      = (u16(*)[HD])      (smem);
    float (*Sc)[Sk]      = (float(*)[Sk])    (smem + 4096);
    u16   (*Pb)[Sk]      = (u16(*)[Sk])      (smem + 4096 + 65536);
    u16   (*Vs)[Sq][HD]  = (u16(*)[Sq][HD])  (smem + 4096 + 65536 + 32768);
    float (*red)[8]      = (float(*)[8])     (smem + 4096 + 65536 + 32768 + 8192);

    const int tid = threadIdx.x, lane = tid & 31, wid = tid >> 5;
    const int qb = blockIdx.x, h = blockIdx.y, b = blockIdx.z;
    const int s0 = qb * Sq;
    const size_t baseQ = (size_t)b * Sk * TD;

    // V staging helper (1 async b128 per thread per 32-row tile)
    const int vr = tid >> 3, vc = (tid & 7) * 8;
    const u16* gV = qkv + baseQ + 2 * Dm + h * HD;
    auto stageV = [&](int n, int buf) {
#if USE_ASYNC_LDS
        __builtin_amdgcn_global_load_async_to_lds_b128(
            (g_i4*)(gV + (size_t)(n + vr) * TD + vc),
            (l_i4*)&Vs[buf][vr][vc], 0, 0);
#else
        *(uint4*)&Vs[buf][vr][vc] =
            *(const uint4*)(gV + (size_t)(n + vr) * TD + vc);
#endif
    };

    // stage Q block [32 x 64]
    {
        int r = tid >> 3, c = (tid & 7) * 8;
        const uint4* g = (const uint4*)(qkv + baseQ + (size_t)(s0 + r) * TD + h * HD + c);
        *(uint4*)&Qs[r][c] = g[0];
    }
    __syncthreads();

    const int lrow = lane & 15, hi = lane >> 4, kc0 = hi * 8;
    const int ti = wid & 1;                        // query tile 0/1

    // ----- phase 1: scores = Q K^T / 8 -----
    for (int j = 0; j < 8; j++) {
        int kt = (wid >> 1) + j * 4;               // key tile 0..31
        v8f acc = {};
        #pragma unroll
        for (int kc = 0; kc < 2; kc++) {           // hd = 64 -> 2 K-steps
            FragBF a, bb;
            int qr = ti * 16 + lrow;
            a.q[0] = *(const uint4*)&Qs[qr][kc * 32 + kc0];
            a.q[1] = *(const uint4*)&Qs[qr][kc * 32 + kc0 + 16];
            int kr = kt * 16 + lrow;               // absolute key index
            const u16* kg = qkv + baseQ + (size_t)kr * TD + Dm + h * HD + kc * 32;
            bb.q[0] = *(const uint4*)(kg + kc0);
            bb.q[1] = *(const uint4*)(kg + kc0 + 16);
            acc = __builtin_amdgcn_wmma_f32_16x16x32_bf16(
                false, a.v, false, bb.v, (short)0, acc, false, false);
        }
        #pragma unroll
        for (int r = 0; r < 8; r++)
            Sc[ti * 16 + hi * 8 + r][kt * 16 + lrow] = acc[r] * 0.125f;
    }

#if USE_ASYNC_LDS
    // kick off first two V tiles; the async engine fills them under softmax
    stageV(0, 0);
    stageV(32, 1);
#endif
    __syncthreads();

    // ----- phase 2: row softmax, emit bf16 probabilities -----
    {
        int row = tid >> 3, sub = tid & 7;
        float mx = -1e30f;
        for (int c = sub; c < Sk; c += 8) mx = fmaxf(mx, Sc[row][c]);
        red[row][sub] = mx; __syncthreads();
        if (sub == 0) {
            float m = red[row][0];
            for (int i = 1; i < 8; i++) m = fmaxf(m, red[row][i]);
            red[row][0] = m;
        }
        __syncthreads();
        float m = red[row][0]; __syncthreads();
        float sm = 0.0f;
        for (int c = sub; c < Sk; c += 8) {
            float e = __expf(Sc[row][c] - m);
            Sc[row][c] = e; sm += e;
        }
        red[row][sub] = sm; __syncthreads();
        if (sub == 0) {
            float s = 0.0f;
            for (int i = 0; i < 8; i++) s += red[row][i];
            red[row][0] = s;
        }
        __syncthreads();
        float inv = 1.0f / red[row][0];
        for (int c = sub; c < Sk; c += 8) Pb[row][c] = f2bf(Sc[row][c] * inv);
    }
    __syncthreads();

    // ----- phase 3: O = P @ V (double-buffered V tiles) -----
    const int tj = wid >> 1;                       // output d-tile 0..3
    v8f oacc = {};
    for (int t = 0; t < 16; t++) {                 // 16 K-steps of 32
        const int n = t * 32;
        const int buf = t & 1;
#if USE_ASYNC_LDS
        if (t < 15) __builtin_amdgcn_s_wait_asynccnt(1);
        else        __builtin_amdgcn_s_wait_asynccnt(0);
#else
        stageV(n, buf);
#endif
        __syncthreads();
        FragBF a, bb;
        int qr = ti * 16 + lrow;
        a.q[0] = *(const uint4*)&Pb[qr][n + kc0];
        a.q[1] = *(const uint4*)&Pb[qr][n + kc0 + 16];
        int dc = tj * 16 + lrow;
        #pragma unroll
        for (int jj = 0; jj < 8; jj++) {
            bb.e[jj]     = Vs[buf][kc0 + jj][dc];
            bb.e[8 + jj] = Vs[buf][kc0 + 16 + jj][dc];
        }
        oacc = __builtin_amdgcn_wmma_f32_16x16x32_bf16(
            false, a.v, false, bb.v, (short)0, oacc, false, false);
        __syncthreads();
#if USE_ASYNC_LDS
        if (t + 2 < 16) stageV((t + 2) * 32, buf); // refill consumed buffer
#endif
    }
    #pragma unroll
    for (int r = 0; r < 8; r++) {
        int srow = s0 + ti * 16 + hi * 8 + r;
        out[((size_t)b * Sk + srow) * Dm + h * HD + tj * 16 + lrow] = oacc[r];
    }
}

// ---------------------------------------------------------------------------
// y = LayerNorm(x + r) * g + b, written fp32 (in-place safe) and bf16
// ---------------------------------------------------------------------------
__global__ __launch_bounds__(256)
void add_ln_kernel(const float* __restrict__ x, const float* __restrict__ r,
                   const float* __restrict__ g, const float* __restrict__ bb,
                   float* __restrict__ y, u16* __restrict__ yh) {
    __shared__ float sred[256];
    const int row = blockIdx.x, tid = threadIdx.x;
    const size_t base = (size_t)row * 768;
    float v[3];
    float s = 0.0f;
    #pragma unroll
    for (int i = 0; i < 3; i++) {
        v[i] = x[base + tid + i * 256] + r[base + tid + i * 256];
        s += v[i];
    }
    sred[tid] = s; __syncthreads();
    for (int st = 128; st > 0; st >>= 1) { if (tid < st) sred[tid] += sred[tid + st]; __syncthreads(); }
    float mu = sred[0] * (1.0f / 768.0f); __syncthreads();
    s = 0.0f;
    #pragma unroll
    for (int i = 0; i < 3; i++) { float d = v[i] - mu; s += d * d; }
    sred[tid] = s; __syncthreads();
    for (int st = 128; st > 0; st >>= 1) { if (tid < st) sred[tid] += sred[tid + st]; __syncthreads(); }
    float inv = rsqrtf(sred[0] * (1.0f / 768.0f) + 1e-5f);
    #pragma unroll
    for (int i = 0; i < 3; i++) {
        int d = tid + i * 256;
        float o = (v[i] - mu) * inv * g[d] + bb[d];
        y[base + d]  = o;
        yh[base + d] = f2bf(o);
    }
}

// ---------------------------------------------------------------------------
// Pooling head pieces
// ---------------------------------------------------------------------------
__global__ __launch_bounds__(256)
void rowdot_kernel(const float* __restrict__ xt, const float* __restrict__ w,
                   float* __restrict__ sout) {    // grid = B*S
    __shared__ float sred[256];
    const int row = blockIdx.x, tid = threadIdx.x;
    const size_t base = (size_t)row * 768;
    float s = 0.0f;
    #pragma unroll
    for (int i = 0; i < 3; i++) { int d = tid + i * 256; s += xt[base + d] * w[d]; }
    sred[tid] = s; __syncthreads();
    for (int st = 128; st > 0; st >>= 1) { if (tid < st) sred[tid] += sred[tid + st]; __syncthreads(); }
    if (tid == 0) sout[row] = sred[0];
}

__global__ __launch_bounds__(256)
void poolsoftmax_kernel(float* __restrict__ sc) {  // grid = B, softmax over S=512
    __shared__ float sred[256];
    const int b = blockIdx.x, tid = threadIdx.x;
    float v0 = sc[b * 512 + tid], v1 = sc[b * 512 + 256 + tid];
    sred[tid] = fmaxf(v0, v1); __syncthreads();
    for (int st = 128; st > 0; st >>= 1) { if (tid < st) sred[tid] = fmaxf(sred[tid], sred[tid + st]); __syncthreads(); }
    float m = sred[0]; __syncthreads();
    float e0 = __expf(v0 - m), e1 = __expf(v1 - m);
    sred[tid] = e0 + e1; __syncthreads();
    for (int st = 128; st > 0; st >>= 1) { if (tid < st) sred[tid] += sred[tid + st]; __syncthreads(); }
    float inv = 1.0f / sred[0];
    sc[b * 512 + tid]       = e0 * inv;
    sc[b * 512 + 256 + tid] = e1 * inv;
}

__global__ __launch_bounds__(256)
void pool_kernel(const float* __restrict__ xt, const float* __restrict__ attn,
                 const float* __restrict__ feats, float* __restrict__ enc) {
    const int b = blockIdx.x, tid = threadIdx.x;   // grid = B
    float acc[3] = {0.f, 0.f, 0.f};
    for (int s = 0; s < 512; s++) {
        float a = attn[b * 512 + s];
        size_t base = ((size_t)b * 512 + s) * 768;
        #pragma unroll
        for (int i = 0; i < 3; i++) acc[i] += a * xt[base + tid + i * 256];
    }
    #pragma unroll
    for (int i = 0; i < 3; i++) enc[b * 784 + tid + i * 256] = acc[i];
    if (tid < 16) enc[b * 784 + 768 + tid] = feats[b * 16 + tid];
}

__global__ __launch_bounds__(256)
void head_kernel(const float* __restrict__ enc, const float* __restrict__ lfc,
                 float* __restrict__ out) {        // grid = B, N = 256, K = 784
    const int b = blockIdx.x, n = threadIdx.x;
    const float* e = enc + b * 784;
    const float* w = lfc + (size_t)n * 784;
    float acc = 0.0f;
    for (int k = 0; k < 784; k++) acc += e[k] * w[k];
    out[b * 256 + n] = tanhf(acc);
}

// ---------------------------------------------------------------------------
extern "C" void kernel_launch(void* const* d_in, const int* in_sizes, int n_in,
                              void* d_out, int out_size, void* d_ws, size_t ws_size,
                              hipStream_t stream) {
    (void)in_sizes; (void)n_in; (void)out_size; (void)ws_size;
    const int B = 32, S = 512, D = 768, H = 12, L = 4, NF = 16;
    const int M = B * S, TD = 3 * D, K = D;

    const float* x     = (const float*)d_in[0];
    const float* feats = (const float*)d_in[1];
    const float* wq    = (const float*)d_in[2];
    const float* wk    = (const float*)d_in[3];
    const float* wv    = (const float*)d_in[4];
    const float* ln1_g = (const float*)d_in[5];
    const float* ln1_b = (const float*)d_in[6];
    const float* ff_w1 = (const float*)d_in[7];
    const float* ff_b1 = (const float*)d_in[8];
    const float* ff_w2 = (const float*)d_in[9];
    const float* ff_b2 = (const float*)d_in[10];
    const float* ln2_g = (const float*)d_in[11];
    const float* ln2_b = (const float*)d_in[12];
    const float* lt_w  = (const float*)d_in[13];
    const float* lt_b  = (const float*)d_in[14];
    const float* agg_w = (const float*)d_in[15];
    const float* lfc_w = (const float*)d_in[16];

    char* ws = (char*)d_ws;
    size_t off = 0;
    auto alloc = [&](size_t bytes) -> char* {
        char* p = ws + off;
        off += (bytes + 255) & ~(size_t)255;
        return p;
    };
    u16*   Wqkv = (u16*)  alloc((size_t)L * TD * D * 2);  // fused qkv weights bf16
    u16*   Wff1 = (u16*)  alloc((size_t)L * D * D * 2);
    u16*   Wff2 = (u16*)  alloc((size_t)L * D * D * 2);
    u16*   Wlt  = (u16*)  alloc((size_t)D * D * 2);
    float* Xf   = (float*)alloc((size_t)M * D * 4);       // activation fp32
    u16*   Xh   = (u16*)  alloc((size_t)M * D * 2);       // activation bf16
    u16*   QKVh = (u16*)  alloc((size_t)M * TD * 2);      // q|k|v bf16
    float* T1f  = (float*)alloc((size_t)M * D * 4);       // attn / ffn2 / xt fp32
    u16*   T2h  = (u16*)  alloc((size_t)M * D * 2);       // ffn1 bf16
    float* Satt = (float*)alloc((size_t)M * 4);           // pooling scores
    float* Enc  = (float*)alloc((size_t)B * (D + NF) * 4);

    // weight conversion (bf16)
    fuse_qkv_kernel<<<(L * TD * D) / 256, 256, 0, stream>>>(wq, wk, wv, Wqkv);
    f32_to_bf16_kernel<<<(L * D * D) / 256, 256, 0, stream>>>(ff_w1, Wff1, L * D * D);
    f32_to_bf16_kernel<<<(L * D * D) / 256, 256, 0, stream>>>(ff_w2, Wff2, L * D * D);
    f32_to_bf16_kernel<<<(D * D) / 256, 256, 0, stream>>>(lt_w, Wlt, D * D);

    posenc_kernel<<<M, 256, 0, stream>>>(x, Xf, Xh);

    const size_t attn_lds = 4096 + 65536 + 32768 + 8192 + 1024;
    for (int l = 0; l < L; l++) {
        // fused QKV projection: [M,768] x [2304,768]^T -> bf16
        gemm_bf16_wmma<0, true><<<dim3(TD / 128, M / 128), 256, 0, stream>>>(
            Xh, Wqkv + (size_t)l * TD * D, nullptr, nullptr, QKVh, M, TD, K);
        // attention -> fp32
        attn_wmma<<<dim3(S / 32, H, B), 256, attn_lds, stream>>>(QKVh, T1f);
        // x = LN(x + attn)
        add_ln_kernel<<<M, 256, 0, stream>>>(Xf, T1f, ln1_g + l * D, ln1_b + l * D, Xf, Xh);
        // FFN1 (relu) -> bf16
        gemm_bf16_wmma<1, true><<<dim3(D / 128, M / 128), 256, 0, stream>>>(
            Xh, Wff1 + (size_t)l * D * D, ff_b1 + l * D, nullptr, T2h, M, D, K);
        // FFN2 -> fp32
        gemm_bf16_wmma<0, false><<<dim3(D / 128, M / 128), 256, 0, stream>>>(
            T2h, Wff2 + (size_t)l * D * D, ff_b2 + l * D, T1f, nullptr, M, D, K);
        // x = LN(x + ffn)
        add_ln_kernel<<<M, 256, 0, stream>>>(Xf, T1f, ln2_g + l * D, ln2_b + l * D, Xf, Xh);
    }

    // xt = tanh(x @ lt_w^T + lt_b)  -> fp32 in T1f
    gemm_bf16_wmma<2, false><<<dim3(D / 128, M / 128), 256, 0, stream>>>(
        Xh, Wlt, lt_b, T1f, nullptr, M, D, K);

    rowdot_kernel<<<M, 256, 0, stream>>>(T1f, agg_w, Satt);
    poolsoftmax_kernel<<<B, 256, 0, stream>>>(Satt);
    pool_kernel<<<B, 256, 0, stream>>>(T1f, Satt, feats, Enc);
    head_kernel<<<B, 256, 0, stream>>>(Enc, lfc_w, (float*)d_out);
}
